// Actor_77695958385088
// MI455X (gfx1250) — matrix operand order repaired and checked by reading, not compiled, not run on hardware
//
#include <hip/hip_runtime.h>
#include <hip/hip_fp16.h>

// ---------------- problem constants (match reference) ----------------
#define BGRAPH 32
#define N0     1584
#define DEG    4
#define E_EDGES (BGRAPH*N0*DEG)     // 202752
#define FIN    32
#define HID    64
#define HEADS  4
#define NH     256                  // HEADS*HID
#define EDIM   8
#define ROUTE  16
#define KP1    1268                 // ceil(0.8*1584)
#define KP2    1015                 // ceil(0.8*1268)
#define KP3    812                  // ceil(0.8*1015)
#define NEGSLOPE 0.2f

typedef _Float16 half_t;
typedef __attribute__((ext_vector_type(16))) _Float16 v16h;
typedef __attribute__((ext_vector_type(8)))  _Float16 v8h;
typedef __attribute__((ext_vector_type(8)))  float    v8f;

// ---------------- small utility kernels ----------------
__global__ void k_cast_f16(const float* __restrict__ s, half_t* __restrict__ d, int n) {
  int i = blockIdx.x * blockDim.x + threadIdx.x;
  if (i < n) d[i] = (half_t)s[i];
}
__global__ void k_fill_f32(float* d, float v, int n) {
  int i = blockIdx.x * blockDim.x + threadIdx.x;
  if (i < n) d[i] = v;
}
__global__ void k_fill_u32(unsigned* d, unsigned v, int n) {
  int i = blockIdx.x * blockDim.x + threadIdx.x;
  if (i < n) d[i] = v;
}
__global__ void k_init_edges(const int* __restrict__ ei, int* s, int* d, int* m, int ne) {
  int i = blockIdx.x * blockDim.x + threadIdx.x;
  if (i < ne) { s[i] = ei[i]; d[i] = ei[E_EDGES + i]; m[i] = 1; }
}

// monotone float<->uint encoding for atomic max over signed floats
__device__ __forceinline__ unsigned fenc(float f) {
  unsigned u = __float_as_uint(f);
  return (u & 0x80000000u) ? ~u : (u | 0x80000000u);
}
__device__ __forceinline__ float fdec(unsigned u) {
  unsigned v = (u & 0x80000000u) ? (u & 0x7FFFFFFFu) : ~u;
  return __uint_as_float(v);
}

// ---------------- WMMA GEMM: C[M,256] = A[M,K](f16) @ B[K,256](f16) + bias ----------------
// One wave32 per 16(M)x64(N) tile. Per K-step: A fragment + all 4 B fragments are
// staged into distinct registers first (one load clause, one wait), then 4
// back-to-back v_wmma ops (A reused from registers; XDL-friendly issue pattern).
// K in {32,64}, M multiple of 16.
// Layouts per CDNA5 ISA 7.12.2:
//   A 16x32 f16: lane l -> row m0+(l&15); elems 0..7 = K kb+8*(l>>4)+0..7,
//                                         elems 8..15 = K kb+16+8*(l>>4)+0..7
//   B 32x16 f16: lane l -> N col n0+(l&15); elems same K order as A
//   C/D f32:     elem r at lane l -> (m0 + r + 8*(l>>4), n0 + (l&15))
__global__ __launch_bounds__(256) void k_gemm_wmma(
    const half_t* __restrict__ A, const half_t* __restrict__ Bw,
    const float* __restrict__ bias, float* __restrict__ C, int M, int K) {
  int wid  = (blockIdx.x * blockDim.x + threadIdx.x) >> 5;
  int lane = threadIdx.x & 31;
  const int tilesN = NH / 64;                 // 4 column-tiles of width 64
  int tm = wid / tilesN, tn = wid % tilesN;
  if (tm * 16 >= M) return;                   // wave-uniform exit (EXEC all-1 for WMMA)
  int m0 = tm * 16, n0 = tn * 64;
  int l15 = lane & 15, hf = lane >> 4;

  v8f acc[4] = {{}, {}, {}, {}};
  for (int kb = 0; kb < K; kb += 32) {
    // --- stage A fragment ---
    const half_t* ap = A + (size_t)(m0 + l15) * K + kb + hf * 8;
    v8h a0 = *(const v8h*)(ap);
    v8h a1 = *(const v8h*)(ap + 16);
    v16h av;
#pragma unroll
    for (int i = 0; i < 8; i++) { av[i] = a0[i]; av[i + 8] = a1[i]; }

    // --- stage all 4 B fragments (distinct registers -> single load clause) ---
    const half_t* bp = Bw + (size_t)(kb + hf * 8) * NH + n0 + l15;
    v16h bv[4];
#pragma unroll
    for (int t = 0; t < 4; t++) {
#pragma unroll
      for (int i = 0; i < 8; i++) { bv[t][i] = bp[(size_t)i * NH + 16 * t]; bv[t][i + 8] = bp[(size_t)(16 + i) * NH + 16 * t]; }
    }

    // --- 4 back-to-back WMMAs reusing the A registers ---
#pragma unroll
    for (int t = 0; t < 4; t++) {
      acc[t] = __builtin_amdgcn_wmma_f32_16x16x32_f16(
          /*neg_a=*/false, av, /*neg_b=*/false, bv[t],
          /*c_mod=*/(short)0, acc[t], /*reuse_a=*/false, /*reuse_b=*/false);
    }
  }
#pragma unroll
  for (int t = 0; t < 4; t++) {
#pragma unroll
    for (int r = 0; r < 8; r++) {
      int m = m0 + r + 8 * hf;
      int n = n0 + 16 * t + l15;
      C[(size_t)m * NH + n] = acc[t][r] + bias[n];
    }
  }
}

// ---------------- GATv2 edge attention: logits + segment max ----------------
// one wave32 per edge; lane handles 8 contiguous channels (head = lane>>3)
__global__ __launch_bounds__(256) void k_edge_logits(
    const float* __restrict__ xl, const float* __restrict__ xr,
    const float* __restrict__ eattr, const float* __restrict__ We,
    const float* __restrict__ att,
    const int* __restrict__ src, const int* __restrict__ dst,
    const int* __restrict__ mask,
    float* __restrict__ logits, unsigned* __restrict__ mxb, int ne) {
  int wid  = (blockIdx.x * blockDim.x + threadIdx.x) >> 5;
  int lane = threadIdx.x & 31;
  if (wid >= ne) return;
  if (!mask[wid]) return;
  int s = src[wid], d = dst[wid];
  float ea[EDIM];
#pragma unroll
  for (int j = 0; j < EDIM; j++) ea[j] = eattr[(size_t)wid * EDIM + j];
  int c0 = lane * 8;
  float part = 0.f;
#pragma unroll
  for (int i = 0; i < 8; i++) {
    int c = c0 + i;
    float ee = 0.f;
#pragma unroll
    for (int j = 0; j < EDIM; j++) ee += ea[j] * We[j * NH + c];
    float v = xl[(size_t)s * NH + c] + xr[(size_t)d * NH + c] + ee;
    v = v > 0.f ? v : NEGSLOPE * v;           // leaky_relu
    part += v * att[c];                       // att flat index == channel
  }
  part += __shfl_xor(part, 1);
  part += __shfl_xor(part, 2);
  part += __shfl_xor(part, 4);
  if ((lane & 7) == 0) {
    int h = lane >> 3;
    logits[(size_t)wid * HEADS + h] = part;
    atomicMax(mxb + (size_t)d * HEADS + h, fenc(part));
  }
}

// exp(logit - segmax) and segment sum of denominators
__global__ void k_edge_exp(const float* __restrict__ logits, const int* __restrict__ mask,
                           const int* __restrict__ dst, const unsigned* __restrict__ mxb,
                           float* __restrict__ den, float* __restrict__ exb, int ne) {
  int t = blockIdx.x * blockDim.x + threadIdx.x;
  if (t >= ne * HEADS) return;
  int e = t >> 2, h = t & 3;
  if (!mask[e]) { exb[t] = 0.f; return; }
  int d = dst[e];
  float mx = fdec(mxb[(size_t)d * HEADS + h]);
  if (!(mx > -3.0e38f && mx < 3.0e38f)) mx = 0.f;
  float ev = __expf(logits[t] - mx);
  exb[t] = ev;
  atomicAdd(den + (size_t)d * HEADS + h, ev);
}

// aggregate alpha * xl[src] into agg[dst] (one wave per edge)
__global__ __launch_bounds__(256) void k_edge_aggregate(
    const float* __restrict__ exb, const float* __restrict__ den,
    const float* __restrict__ xl, const int* __restrict__ src,
    const int* __restrict__ dst, const int* __restrict__ mask,
    float* __restrict__ agg, int ne) {
  int wid  = (blockIdx.x * blockDim.x + threadIdx.x) >> 5;
  int lane = threadIdx.x & 31;
  if (wid >= ne) return;
  if (!mask[wid]) return;
  int s = src[wid], d = dst[wid];
  int h = lane >> 3;
  float dn    = den[(size_t)d * HEADS + h];
  float alpha = exb[(size_t)wid * HEADS + h] / fmaxf(dn, 1e-16f);
  int c0 = lane * 8;
#pragma unroll
  for (int i = 0; i < 8; i++) {
    int c = c0 + i;
    atomicAdd(agg + (size_t)d * NH + c, alpha * xl[(size_t)s * NH + c]);
  }
}

// head-mean + bias + relu  -> node features [nodes, HID]
__global__ void k_node_finalize(const float* __restrict__ agg, const float* __restrict__ bo,
                                float* __restrict__ h, int nodes) {
  int t = blockIdx.x * blockDim.x + threadIdx.x;
  if (t >= nodes * HID) return;
  int n = t / HID, c = t % HID;
  const float* a = agg + (size_t)n * NH;
  float v = 0.25f * (a[c] + a[HID + c] + a[2 * HID + c] + a[3 * HID + c]) + bo[c];
  h[t] = fmaxf(v, 0.f);
}

// ---------------- TopK pooling ----------------
__global__ void k_scores(const float* __restrict__ h, const float* __restrict__ w,
                         float* __restrict__ s, int nodes) {
  int n = blockIdx.x * blockDim.x + threadIdx.x;
  if (n >= nodes) return;
  float nrm = 0.f;
#pragma unroll
  for (int j = 0; j < HID; j++) nrm += w[j] * w[j];
  nrm = sqrtf(nrm);
  float dot = 0.f;
#pragma unroll
  for (int j = 0; j < HID; j++) dot += h[(size_t)n * HID + j] * w[j];
  s[n] = tanhf(dot / nrm);
}

// exact top-k per graph via O(n^2) rank (tie-break by index == lax.top_k)
__global__ __launch_bounds__(256) void k_topk(const float* __restrict__ s, int n_per, int k,
                                              int* __restrict__ new_id, int* __restrict__ perm) {
  __shared__ float sc[N0];
  __shared__ int   nid[N0];
  int g = blockIdx.x, tid = threadIdx.x;
  for (int i = tid; i < n_per; i += blockDim.x) sc[i] = s[g * n_per + i];
  __syncthreads();
  for (int i = tid; i < n_per; i += blockDim.x) {
    float b = sc[i];
    int rank = 0;
    for (int j = 0; j < n_per; j++) {
      float a = sc[j];
      rank += (a > b) || (a == b && j < i);
    }
    nid[i] = (rank < k) ? 1 : 0;
  }
  __syncthreads();
  if (tid == 0) {
    int pos = 0;
    for (int i = 0; i < n_per; i++) { int kk = nid[i]; nid[i] = kk ? pos : -1; pos += kk; }
  }
  __syncthreads();
  for (int i = tid; i < n_per; i += blockDim.x) {
    int p = nid[i];
    new_id[g * n_per + i] = (p >= 0) ? (g * k + p) : -1;
    if (p >= 0) perm[g * k + p] = g * n_per + i;
  }
}

__global__ void k_pool_gather(const float* __restrict__ h, const float* __restrict__ s,
                              const int* __restrict__ perm, float* __restrict__ out, int nnew) {
  int t = blockIdx.x * blockDim.x + threadIdx.x;
  if (t >= nnew * HID) return;
  int nn = t / HID, c = t % HID;
  int old = perm[nn];
  out[t] = h[(size_t)old * HID + c] * s[old];
}

__global__ void k_edge_remap(int* src, int* dst, int* mask, const int* __restrict__ new_id, int ne) {
  int e = blockIdx.x * blockDim.x + threadIdx.x;
  if (e >= ne) return;
  int s = new_id[src[e]], d = new_id[dst[e]];
  int m = mask[e] && (s >= 0) && (d >= 0);
  src[e] = m ? s : 0;
  dst[e] = m ? d : 0;
  mask[e] = m;
}

// readout (mean||max per graph) accumulated into r (r = x1+x2+x3 progressively)
__global__ void k_readout(const float* __restrict__ h, int k, float* __restrict__ r) {
  int t = blockIdx.x * blockDim.x + threadIdx.x;
  if (t >= BGRAPH * 2 * HID) return;
  int g = t / (2 * HID), cc = t % (2 * HID);
  const float* hp = h + (size_t)g * k * HID;
  if (cc < HID) {
    float sum = 0.f;
    for (int i = 0; i < k; i++) sum += hp[(size_t)i * HID + cc];
    r[t] += sum / (float)k;
  } else {
    int c = cc - HID;
    float mx = -3.4e38f;
    for (int i = 0; i < k; i++) mx = fmaxf(mx, hp[(size_t)i * HID + c]);
    r[t] += mx;
  }
}

// ---------------- final MLP + softmax (tiny: 32x128 -> 16) ----------------
__global__ __launch_bounds__(256) void k_mlp(const float* __restrict__ r,
    const float* __restrict__ fc1W, const float* __restrict__ fc1b,
    const float* __restrict__ fc2W, const float* __restrict__ fc2b,
    const float* __restrict__ fc3W, const float* __restrict__ fc3b,
    float* __restrict__ out) {
  __shared__ float h0[BGRAPH * 2 * HID];
  __shared__ float h1[BGRAPH * HID];
  __shared__ float h2[BGRAPH * HID];
  __shared__ float h3[BGRAPH * ROUTE];
  int tid = threadIdx.x;
  for (int i = tid; i < BGRAPH * 2 * HID; i += blockDim.x) h0[i] = r[i];
  __syncthreads();
  for (int o = tid; o < BGRAPH * HID; o += blockDim.x) {
    int row = o / HID, c = o % HID;
    float acc = fc1b[c];
    for (int j = 0; j < 2 * HID; j++) acc += h0[row * 2 * HID + j] * fc1W[j * HID + c];
    h1[o] = fmaxf(acc, 0.f);
  }
  __syncthreads();
  for (int o = tid; o < BGRAPH * HID; o += blockDim.x) {
    int row = o / HID, c = o % HID;
    float acc = fc2b[c];
    for (int j = 0; j < HID; j++) acc += h1[row * HID + j] * fc2W[j * HID + c];
    h2[o] = fmaxf(acc, 0.f);
  }
  __syncthreads();
  for (int o = tid; o < BGRAPH * ROUTE; o += blockDim.x) {
    int row = o / ROUTE, c = o % ROUTE;
    float acc = fc3b[c];
    for (int j = 0; j < HID; j++) acc += h2[row * HID + j] * fc3W[j * ROUTE + c];
    h3[o] = acc;
  }
  __syncthreads();
  if (tid < BGRAPH) {
    float mx = -3.4e38f;
    for (int q = 0; q < ROUTE; q++) mx = fmaxf(mx, h3[tid * ROUTE + q]);
    float sum = 0.f, ex[ROUTE];
    for (int q = 0; q < ROUTE; q++) { ex[q] = __expf(h3[tid * ROUTE + q] - mx); sum += ex[q]; }
    for (int q = 0; q < ROUTE; q++) out[tid * ROUTE + q] = ex[q] / sum;
  }
}

// ---------------- launcher ----------------
extern "C" void kernel_launch(void* const* d_in, const int* in_sizes, int n_in,
                              void* d_out, int out_size, void* d_ws, size_t ws_size,
                              hipStream_t stream) {
  (void)in_sizes; (void)n_in; (void)out_size; (void)ws_size;
  const float* x      = (const float*)d_in[0];
  const float* eattr  = (const float*)d_in[1];
  const int*   eindex = (const int*)d_in[2];
  const float* gWl[3] = {(const float*)d_in[4],  (const float*)d_in[11], (const float*)d_in[18]};
  const float* gbl[3] = {(const float*)d_in[5],  (const float*)d_in[12], (const float*)d_in[19]};
  const float* gWr[3] = {(const float*)d_in[6],  (const float*)d_in[13], (const float*)d_in[20]};
  const float* gbr[3] = {(const float*)d_in[7],  (const float*)d_in[14], (const float*)d_in[21]};
  const float* gWe[3] = {(const float*)d_in[8],  (const float*)d_in[15], (const float*)d_in[22]};
  const float* gat[3] = {(const float*)d_in[9],  (const float*)d_in[16], (const float*)d_in[23]};
  const float* gbo[3] = {(const float*)d_in[10], (const float*)d_in[17], (const float*)d_in[24]};
  const float* pw[3]  = {(const float*)d_in[25], (const float*)d_in[26], (const float*)d_in[27]};
  const float* fc1W = (const float*)d_in[28]; const float* fc1b = (const float*)d_in[29];
  const float* fc2W = (const float*)d_in[30]; const float* fc2b = (const float*)d_in[31];
  const float* fc3W = (const float*)d_in[32]; const float* fc3b = (const float*)d_in[33];

  const int M1 = BGRAPH * N0;                 // 50688 (multiple of 16)
  // ---- workspace arena (~148 MB), 256B aligned ----
  char* p = (char*)d_ws;
  auto alloc = [&](size_t bytes) -> void* {
    void* r = (void*)p; p += (bytes + 255) & ~(size_t)255; return r;
  };
  float*    xl   = (float*)alloc((size_t)M1 * NH * 4);
  float*    xr   = (float*)alloc((size_t)M1 * NH * 4);   // reused as agg buffer
  float*    hA   = (float*)alloc((size_t)M1 * HID * 4);
  float*    hB   = (float*)alloc((size_t)M1 * HID * 4);
  half_t*   abuf = (half_t*)alloc((size_t)M1 * HID * 2);
  half_t*   wlb  = (half_t*)alloc((size_t)HID * NH * 2);
  half_t*   wrb  = (half_t*)alloc((size_t)HID * NH * 2);
  float*    lgt  = (float*)alloc((size_t)E_EDGES * HEADS * 4);
  float*    exb  = (float*)alloc((size_t)E_EDGES * HEADS * 4);
  unsigned* mxb  = (unsigned*)alloc((size_t)M1 * HEADS * 4);
  float*    den  = (float*)alloc((size_t)M1 * HEADS * 4);
  float*    sbuf = (float*)alloc((size_t)M1 * 4);
  int*      nidb = (int*)alloc((size_t)M1 * 4);
  int*      perm = (int*)alloc((size_t)M1 * 4);
  int*      csrc = (int*)alloc((size_t)E_EDGES * 4);
  int*      cdst = (int*)alloc((size_t)E_EDGES * 4);
  int*      cmsk = (int*)alloc((size_t)E_EDGES * 4);
  float*    racc = (float*)alloc((size_t)BGRAPH * 2 * HID * 4);

  auto cdiv = [](long a, long b) { return (int)((a + b - 1) / b); };
  const int T = 256;

  k_init_edges<<<cdiv(E_EDGES, T), T, 0, stream>>>(eindex, csrc, cdst, cmsk, E_EDGES);
  k_fill_f32<<<cdiv(BGRAPH * 2 * HID, T), T, 0, stream>>>(racc, 0.f, BGRAPH * 2 * HID);

  const float* hin = x;
  int Kin = FIN, M = M1;
  const int nper[3] = {N0, KP1, KP2};
  const int kpk[3]  = {KP1, KP2, KP3};
  float* hout  = hA;
  float* hpool = hB;

  for (int L = 0; L < 3; L++) {
    // dense transforms via WMMA (xl = x@Wl+bl, xr = x@Wr+br), f16 inputs, f32 accum
    k_cast_f16<<<cdiv((long)M * Kin, T), T, 0, stream>>>(hin, abuf, M * Kin);
    k_cast_f16<<<cdiv(Kin * NH, T), T, 0, stream>>>(gWl[L], wlb, Kin * NH);
    k_cast_f16<<<cdiv(Kin * NH, T), T, 0, stream>>>(gWr[L], wrb, Kin * NH);
    int tiles = (M / 16) * (NH / 64);          // 16x64 tile per wave
    k_gemm_wmma<<<cdiv(tiles, 8), T, 0, stream>>>(abuf, wlb, gbl[L], xl, M, Kin);
    k_gemm_wmma<<<cdiv(tiles, 8), T, 0, stream>>>(abuf, wrb, gbr[L], xr, M, Kin);

    // attention: logits + segment max -> exp/denominator -> weighted aggregation
    k_fill_u32<<<cdiv(M * HEADS, T), T, 0, stream>>>(mxb, 0x007FFFFFu, M * HEADS); // enc(-inf)
    k_fill_f32<<<cdiv(M * HEADS, T), T, 0, stream>>>(den, 0.f, M * HEADS);
    k_edge_logits<<<cdiv(E_EDGES, 8), T, 0, stream>>>(xl, xr, eattr, gWe[L], gat[L],
                                                      csrc, cdst, cmsk, lgt, mxb, E_EDGES);
    k_edge_exp<<<cdiv((long)E_EDGES * HEADS, T), T, 0, stream>>>(lgt, cmsk, cdst, mxb, den, exb, E_EDGES);
    k_fill_f32<<<cdiv((long)M * NH, T), T, 0, stream>>>(xr, 0.f, M * NH);  // xr reused as agg
    k_edge_aggregate<<<cdiv(E_EDGES, 8), T, 0, stream>>>(exb, den, xl, csrc, cdst, cmsk, xr, E_EDGES);
    k_node_finalize<<<cdiv((long)M * HID, T), T, 0, stream>>>(xr, gbo[L], hout, M);

    // top-k pooling + edge remap + readout accumulation
    k_scores<<<cdiv(M, T), T, 0, stream>>>(hout, pw[L], sbuf, M);
    k_topk<<<BGRAPH, T, 0, stream>>>(sbuf, nper[L], kpk[L], nidb, perm);
    int Mn = BGRAPH * kpk[L];
    k_pool_gather<<<cdiv((long)Mn * HID, T), T, 0, stream>>>(hout, sbuf, perm, hpool, Mn);
    k_edge_remap<<<cdiv(E_EDGES, T), T, 0, stream>>>(csrc, cdst, cmsk, nidb, E_EDGES);
    k_readout<<<cdiv(BGRAPH * 2 * HID, T), T, 0, stream>>>(hpool, kpk[L], racc);

    hin = hpool; M = Mn; Kin = HID;
    float* tmp = hout; hout = hpool; hpool = tmp;
  }

  k_mlp<<<1, T, 0, stream>>>(racc, fc1W, fc1b, fc2W, fc2b, fc3W, fc3b, (float*)d_out);
}
